// NeuralTuringMachine_7473243095502
// MI455X (gfx1250) — matrix operand order repaired and checked by reading, not compiled, not run on hardware
//
#include <hip/hip_runtime.h>
#include <math.h>

typedef __attribute__((ext_vector_type(16))) _Float16 v16h;
typedef __attribute__((ext_vector_type(8)))  float    v8f;

#define NTM_B   128
#define NTM_IN  256
#define NTM_H   512
#define NTM_OUT 256
#define NTM_M   64
#define NTM_N   65536
#define NTM_CIN 320          // IN + NR*M
#define NTM_4H  2048
#define NTM_P   258
#define NTM_NH  514          // OUT + P combined head-GEMM width

// ---------------------------------------------------------------------------
// Small helpers
// ---------------------------------------------------------------------------
__device__ __forceinline__ float sigmoidf_(float x) { return 1.0f / (1.0f + __expf(-x)); }
__device__ __forceinline__ float softplusf_(float x) {
  return (x > 20.0f) ? x : log1pf(__expf(x));
}

// A-matrix 16x32 f16 tile (M x K), contiguous K order (matches B):
// lanes 0-15 : elems 0..15 = K 0..15   (row = lane)
// lanes 16-31: elems 0..15 = K 16..31  (row = lane-16)
__device__ __forceinline__ v16h load_a16x32(const float* __restrict__ base,
                                            int row0, int ld, int k0, int lane) {
  int r  = row0 + (lane & 15);
  int kh = (lane >> 4) * 16;
  const float* p = base + (size_t)r * ld + k0 + kh;
  v16h o;
#pragma unroll
  for (int j = 0; j < 8; ++j) o[j]     = (_Float16)p[j];
#pragma unroll
  for (int j = 0; j < 8; ++j) o[8 + j] = (_Float16)p[8 + j];
  return o;
}

// B-matrix 32x16 f16 tile (K x N), B[k][n] = W[n][k] (W row-major, K contig):
// lanes 0-15 (n = lane)    hold K 0..15  across the 8 VGPRs (pairs)
// lanes 16-31 (n = lane-16) hold K 16..31
__device__ __forceinline__ v16h load_b32x16(const float* __restrict__ Wrow,
                                            int k0, int lane) {
  const float* p = Wrow + k0 + (lane >> 4) * 16;
  v16h o;
#pragma unroll
  for (int j = 0; j < 16; ++j) o[j] = (_Float16)p[j];
  return o;
}

// ---------------------------------------------------------------------------
// 1) ci = concat(x, read_vectors)   [128 x 320]
// ---------------------------------------------------------------------------
__global__ void ntm_concat(const float* __restrict__ x, const float* __restrict__ rv,
                           float* __restrict__ ci) {
  int idx = blockIdx.x * blockDim.x + threadIdx.x;
  if (idx >= NTM_B * NTM_CIN) return;
  int b = idx / NTM_CIN, j = idx - b * NTM_CIN;
  ci[idx] = (j < NTM_IN) ? x[b * NTM_IN + j] : rv[j - NTM_IN];
}

// ---------------------------------------------------------------------------
// 2) gates = ci @ W_ih^T + b_ih + b_hh     [128 x 2048], WMMA f16->f32
//    grid: (2048/16, 128/16), one wave (32 threads) per 16x16 tile
// ---------------------------------------------------------------------------
__global__ void ntm_gemm_gates(const float* __restrict__ ci, const float* __restrict__ Wih,
                               const float* __restrict__ bih, const float* __restrict__ bhh,
                               float* __restrict__ gates) {
  int lane = threadIdx.x;
  int n0 = blockIdx.x * 16;
  int m0 = blockIdx.y * 16;
  v8f c = {0.f, 0.f, 0.f, 0.f, 0.f, 0.f, 0.f, 0.f};
  int nrow = n0 + (lane & 15);
#pragma unroll
  for (int k0 = 0; k0 < NTM_CIN; k0 += 32) {
    v16h a = load_a16x32(ci, m0, NTM_CIN, k0, lane);
    v16h b = load_b32x16(Wih + (size_t)nrow * NTM_CIN, k0, lane);
    c = __builtin_amdgcn_wmma_f32_16x16x32_f16(false, a, false, b, (short)0, c, false, false);
  }
  float bias = bih[nrow] + bhh[nrow];
  int mh = 8 * (lane >> 4);
#pragma unroll
  for (int v = 0; v < 8; ++v) {
    int row = m0 + mh + v;
    gates[(size_t)row * NTM_4H + nrow] = c[v] + bias;
  }
}

// ---------------------------------------------------------------------------
// 3) LSTM activations (cx0 = 0): hx = sig(o) * tanh(sig(i)*tanh(g))
// ---------------------------------------------------------------------------
__global__ void ntm_act(const float* __restrict__ gates, float* __restrict__ hx) {
  int idx = blockIdx.x * blockDim.x + threadIdx.x;
  if (idx >= NTM_B * NTM_H) return;
  int b = idx / NTM_H, h = idx - b * NTM_H;
  const float* g = gates + (size_t)b * NTM_4H;
  float cx = sigmoidf_(g[h]) * tanhf(g[2 * NTM_H + h]);
  hx[idx] = sigmoidf_(g[3 * NTM_H + h]) * tanhf(cx);
}

// ---------------------------------------------------------------------------
// 4) Fused head GEMM over K=512:
//    n <  256 -> ctrl_out = hx@W_out^T + b_out  (straight to d_out)
//    n >= 256 -> head_params = hx@W_p^T + b_p   (ws, ld = 258)
//    grid: (33, 8); N padded 514 -> 528, predicated loads/stores
// ---------------------------------------------------------------------------
__global__ void ntm_gemm_heads(const float* __restrict__ hx,
                               const float* __restrict__ Wout, const float* __restrict__ bout,
                               const float* __restrict__ Wp,   const float* __restrict__ bp,
                               float* __restrict__ ctrl, float* __restrict__ hp) {
  int lane = threadIdx.x;
  int n0 = blockIdx.x * 16;
  int m0 = blockIdx.y * 16;
  int n = n0 + (lane & 15);
  const float* wrow = nullptr;
  if (n < NTM_OUT)      wrow = Wout + (size_t)n * NTM_H;
  else if (n < NTM_NH)  wrow = Wp + (size_t)(n - NTM_OUT) * NTM_H;

  v8f c = {0.f, 0.f, 0.f, 0.f, 0.f, 0.f, 0.f, 0.f};
#pragma unroll
  for (int k0 = 0; k0 < NTM_H; k0 += 32) {
    v16h a = load_a16x32(hx, m0, NTM_H, k0, lane);
    v16h b;
    if (wrow) {
      b = load_b32x16(wrow, k0, lane);
    } else {
#pragma unroll
      for (int j = 0; j < 16; ++j) b[j] = (_Float16)0.0f;
    }
    c = __builtin_amdgcn_wmma_f32_16x16x32_f16(false, a, false, b, (short)0, c, false, false);
  }
  int mh = 8 * (lane >> 4);
  if (n < NTM_OUT) {
    float bias = bout[n];
#pragma unroll
    for (int v = 0; v < 8; ++v) ctrl[(size_t)(m0 + mh + v) * NTM_OUT + n] = c[v] + bias;
  } else if (n < NTM_NH) {
    int nn = n - NTM_OUT;
    float bias = bp[nn];
#pragma unroll
    for (int v = 0; v < 8; ++v) hp[(size_t)(m0 + mh + v) * NTM_P + nn] = c[v] + bias;
  }
}

// ---------------------------------------------------------------------------
// 5) Per-step params: q_t = softplus(beta)*tanh(k)/(||tanh(k)||+1e-8),
//    e_t = sigmoid(.), a_t = tanh(.). Block 128 = read head (batch 0).
//    129 blocks x 64 threads.
// ---------------------------------------------------------------------------
__global__ void ntm_prep(const float* __restrict__ hp, float* __restrict__ q,
                         float* __restrict__ eP, float* __restrict__ aP) {
  __shared__ float s[NTM_M];
  int b = blockIdx.x, c = threadIdx.x;
  float k, beta;
  if (b < NTM_B) {
    const float* p = hp + (size_t)b * NTM_P;
    k = tanhf(p[c]);
    beta = softplusf_(p[NTM_M]);
    eP[b * NTM_M + c] = sigmoidf_(p[NTM_M + 1 + c]);
    aP[b * NTM_M + c] = tanhf(p[2 * NTM_M + 1 + c]);
  } else {  // read head, batch 0
    const float* p = hp + (3 * NTM_M + 1);
    k = tanhf(p[c]);
    beta = softplusf_(p[NTM_M]);
  }
  s[c] = k * k;
  __syncthreads();
  for (int st = 32; st; st >>= 1) { if (c < st) s[c] += s[c + st]; __syncthreads(); }
  float nrm = sqrtf(s[0]);
  q[b * NTM_M + c] = beta * k / (nrm + 1e-8f);
}

// ---------------------------------------------------------------------------
// 6) Fused memory step. 256 blocks x 256 threads, 256 rows/block.
//    doUpdate=1: w_i = exp(score_i - M)/S ; row = row*(1-w*e) + w*a
//    Always: write row to dstMem, score'_i = (row . q)/(||row||+1e-8),
//    per-block online-softmax partials (blockMax, sum of exp(x-blockMax)).
// ---------------------------------------------------------------------------
__global__ void ntm_step(const float* __restrict__ srcMem, float* __restrict__ dstMem,
                         float* __restrict__ scores, const float* __restrict__ red,
                         const float* __restrict__ eP, const float* __restrict__ aP,
                         const float* __restrict__ q,
                         float* __restrict__ blockMax, float* __restrict__ blockSum,
                         int doUpdate) {
  __shared__ float s_sc[256];
  __shared__ float s_red[256];
  int tid  = threadIdx.x;
  int lane = tid & 31;
  int warp = tid >> 5;
  int rowBase = blockIdx.x * 256 + warp * 32;

  float gM = 0.f, invS = 0.f;
  float2 ev = {0.f, 0.f}, av = {0.f, 0.f};
  if (doUpdate) {
    gM = red[0];
    invS = 1.0f / red[1];
    ev = ((const float2*)eP)[lane];
    av = ((const float2*)aP)[lane];
  }
  float2 qv = ((const float2*)q)[lane];

#pragma unroll 4
  for (int r = 0; r < 32; ++r) {
    int row = rowBase + r;
    if (r + 1 < 32)
      __builtin_prefetch(srcMem + (size_t)(row + 1) * NTM_M + 2 * lane, 0, 0);
    float2 mv = ((const float2*)(srcMem + (size_t)row * NTM_M))[lane];
    if (doUpdate) {
      float w = __expf(scores[row] - gM) * invS;
      mv.x = mv.x * (1.0f - w * ev.x) + w * av.x;
      mv.y = mv.y * (1.0f - w * ev.y) + w * av.y;
    }
    ((float2*)(dstMem + (size_t)row * NTM_M))[lane] = mv;
    float dot = mv.x * qv.x + mv.y * qv.y;
    float nrm = mv.x * mv.x + mv.y * mv.y;
#pragma unroll
    for (int off = 16; off; off >>= 1) {
      dot += __shfl_xor(dot, off, 32);
      nrm += __shfl_xor(nrm, off, 32);
    }
    if (lane == 0) {
      float sc = dot / (sqrtf(nrm) + 1e-8f);
      scores[row] = sc;
      s_sc[warp * 32 + r] = sc;
    }
  }
  __syncthreads();
  // block softmax partials over the 256 scores
  float v = s_sc[tid];
  s_red[tid] = v;
  __syncthreads();
  for (int st = 128; st; st >>= 1) {
    if (tid < st) s_red[tid] = fmaxf(s_red[tid], s_red[tid + st]);
    __syncthreads();
  }
  float bm = s_red[0];
  __syncthreads();
  s_red[tid] = __expf(v - bm);
  __syncthreads();
  for (int st = 128; st; st >>= 1) {
    if (tid < st) s_red[tid] += s_red[tid + st];
    __syncthreads();
  }
  if (tid == 0) { blockMax[blockIdx.x] = bm; blockSum[blockIdx.x] = s_red[0]; }
}

// ---------------------------------------------------------------------------
// 7) Combine 256 per-block partials -> global (max M, S = sum exp(x-M))
// ---------------------------------------------------------------------------
__global__ void ntm_reduce(const float* __restrict__ blockMax,
                           const float* __restrict__ blockSum,
                           float* __restrict__ red) {
  __shared__ float sm[256];
  __shared__ float ss[256];
  int tid = threadIdx.x;
  float m = blockMax[tid], s = blockSum[tid];
  sm[tid] = m;
  __syncthreads();
  for (int st = 128; st; st >>= 1) {
    if (tid < st) sm[tid] = fmaxf(sm[tid], sm[tid + st]);
    __syncthreads();
  }
  float gm = sm[0];
  __syncthreads();
  ss[tid] = s * __expf(m - gm);
  __syncthreads();
  for (int st = 128; st; st >>= 1) {
    if (tid < st) ss[tid] += ss[tid + st];
    __syncthreads();
  }
  if (tid == 0) { red[0] = gm; red[1] = ss[0]; }
}

// ---------------------------------------------------------------------------
// 8) Read: per-block partial of sum_i w_i * mem[i][:]  -> readPart[256][64]
// ---------------------------------------------------------------------------
__global__ void ntm_read(const float* __restrict__ mem, const float* __restrict__ scores,
                         const float* __restrict__ red, float* __restrict__ readPart) {
  __shared__ float s_acc[8][NTM_M];
  int tid = threadIdx.x, lane = tid & 31, warp = tid >> 5;
  float gM = red[0], invS = 1.0f / red[1];
  int rowBase = blockIdx.x * 256 + warp * 32;
  float2 acc = {0.f, 0.f};
#pragma unroll 4
  for (int r = 0; r < 32; ++r) {
    int row = rowBase + r;
    float w = __expf(scores[row] - gM) * invS;
    float2 mv = ((const float2*)(mem + (size_t)row * NTM_M))[lane];
    acc.x += w * mv.x;
    acc.y += w * mv.y;
  }
  s_acc[warp][2 * lane]     = acc.x;
  s_acc[warp][2 * lane + 1] = acc.y;
  __syncthreads();
  if (tid < NTM_M) {
    float t = 0.f;
#pragma unroll
    for (int w = 0; w < 8; ++w) t += s_acc[w][tid];
    readPart[blockIdx.x * NTM_M + tid] = t;
  }
}

__global__ void ntm_read_final(const float* __restrict__ readPart, float* __restrict__ out) {
  int c = threadIdx.x;  // 64 threads
  float t = 0.f;
  for (int b = 0; b < 256; ++b) t += readPart[b * NTM_M + c];
  out[c] = t;
}

// ---------------------------------------------------------------------------
// Host launch
// ---------------------------------------------------------------------------
extern "C" void kernel_launch(void* const* d_in, const int* in_sizes, int n_in,
                              void* d_out, int out_size, void* d_ws, size_t ws_size,
                              hipStream_t stream) {
  (void)in_sizes; (void)n_in; (void)out_size; (void)ws_size;
  const float* x    = (const float*)d_in[0];   // [128,256]
  const float* rv   = (const float*)d_in[1];   // [1,64]
  const float* mem0 = (const float*)d_in[2];   // [65536,64]
  const float* Wih  = (const float*)d_in[3];   // [2048,320]
  // d_in[4] = W_hh (unused: hx0 == 0)
  const float* bih  = (const float*)d_in[5];   // [2048]
  const float* bhh  = (const float*)d_in[6];   // [2048]
  const float* Wout = (const float*)d_in[7];   // [256,512]
  const float* bout = (const float*)d_in[8];   // [256]
  const float* Wp   = (const float*)d_in[9];   // [258,512]
  const float* bp   = (const float*)d_in[10];  // [258]

  float* out_ctrl = (float*)d_out;                 // [128,256]
  float* out_read = out_ctrl + NTM_B * NTM_OUT;    // [64]

  // workspace layout (floats)
  float* W = (float*)d_ws;
  size_t off = 0;
  float* mem      = W + off; off += (size_t)NTM_N * NTM_M;   // 16.78 MB
  float* scores   = W + off; off += NTM_N;
  float* bMax     = W + off; off += 256;
  float* bSum     = W + off; off += 256;
  float* red      = W + off; off += 16;
  float* readPart = W + off; off += 256 * NTM_M;
  float* ci       = W + off; off += (size_t)NTM_B * NTM_CIN;
  float* gates    = W + off; off += (size_t)NTM_B * NTM_4H;
  float* hx       = W + off; off += (size_t)NTM_B * NTM_H;
  float* hp       = W + off; off += (size_t)NTM_B * NTM_P;
  float* q        = W + off; off += (size_t)(NTM_B + 1) * NTM_M;  // 128 writes + 1 read
  float* eP       = W + off; off += (size_t)NTM_B * NTM_M;
  float* aP       = W + off; off += (size_t)NTM_B * NTM_M;

  // controller
  ntm_concat<<<(NTM_B * NTM_CIN + 255) / 256, 256, 0, stream>>>(x, rv, ci);
  ntm_gemm_gates<<<dim3(NTM_4H / 16, NTM_B / 16), 32, 0, stream>>>(ci, Wih, bih, bhh, gates);
  ntm_act<<<(NTM_B * NTM_H + 255) / 256, 256, 0, stream>>>(gates, hx);
  ntm_gemm_heads<<<dim3(33, NTM_B / 16), 32, 0, stream>>>(hx, Wout, bout, Wp, bp, out_ctrl, hp);
  ntm_prep<<<NTM_B + 1, NTM_M, 0, stream>>>(hp, q, eP, aP);

  // step 0 scores (and copy memory0 -> ws mem; inputs must not be mutated)
  ntm_step<<<256, 256, 0, stream>>>(mem0, mem, scores, red, eP, aP, q, bMax, bSum, 0);
  ntm_reduce<<<1, 256, 0, stream>>>(bMax, bSum, red);

  // 128 sequential erase/add writes; each fused with next step's cosine scores
  for (int t = 0; t < NTM_B; ++t) {
    const float* qn = q + (size_t)(t + 1) * NTM_M;  // t==127 -> read-head query
    ntm_step<<<256, 256, 0, stream>>>(mem, mem, scores, red,
                                      eP + (size_t)t * NTM_M, aP + (size_t)t * NTM_M,
                                      qn, bMax, bSum, 1);
    ntm_reduce<<<1, 256, 0, stream>>>(bMax, bSum, red);
  }

  // final read: w @ mem  -> [64]
  ntm_read<<<256, 256, 0, stream>>>(mem, scores, red, readPart);
  ntm_read_final<<<1, NTM_M, 0, stream>>>(readPart, out_read);
}
